// WaveRNN_14224931684672
// MI455X (gfx1250) — matrix operand order repaired and checked by reading, not compiled, not run on hardware
//
#include <hip/hip_runtime.h>
#include <hip/hip_bf16.h>

// ---------------------------------------------------------------------------
// WaveRNN-style pipeline for MI455X (gfx1250, wave32, WMMA).
//  K0: pack/convert inputs + weights to bf16, zero GRU state + barrier
//  K1: xp = inp @ w_ih^T + b_ih          (WMMA bf16, f32 out, [T][16][3072])
//  K2: persistent GRU: 64 WGs x 96 thr, 1 WG = 16 hidden units = 3 gate tiles,
//      device-wide atomic barrier per step, double-buffered h state,
//      dual WMMA accumulators to halve the dependent-chain latency
//  K3: fused head: async-LDS stage -> relu(A@w1^T+b1) -> @w2^T+b2 -> log_softmax
// Workspace use ~970 MB (xp fp32 dominates); assumes ws_size is sufficient.
// ---------------------------------------------------------------------------

typedef __attribute__((ext_vector_type(16))) __bf16 v16bf;
typedef __attribute__((ext_vector_type(8)))  __bf16 v8bf;
typedef __attribute__((ext_vector_type(8)))  float  v8f;
typedef int v4i_t __attribute__((vector_size(16)));   // matches async-LDS builtin pointee

#define T_LEN  4096
#define B_SZ   16
#define H_SZ   1024
#define G3H    3072
#define DPAD   160     // 147 input dims padded to 5*32
#define FC_SZ  1024
#define NC_SZ  256
#define KIN    576     // HALF + AUX = 512 + 64 = 18*32
#define GRU_WGS 64

#define GLOBAL_AS __attribute__((address_space(1)))
#define LDS_AS    __attribute__((address_space(3)))

__device__ __forceinline__ __bf16 f2bf(float x) {
  union { float f; unsigned u; } a; a.f = x;
  unsigned r = a.u + 0x7FFFu + ((a.u >> 16) & 1u);   // round-to-nearest-even
  union { unsigned short s; __bf16 b; } o; o.s = (unsigned short)(r >> 16);
  return o.b;
}
__device__ __forceinline__ float bf2f(__bf16 x) {
  union { unsigned short s; __bf16 b; } i; i.b = x;
  union { unsigned u; float f; } o; o.u = ((unsigned)i.s) << 16;
  return o.f;
}

union ABFrag { v16bf v; v8bf h8[2]; };

// A fragment (16x32 bf16, row-major source, row stride = ld elements).
// ISA layout: lanes 0-15 M=0..15 {K: e0..7 -> kb+0..7, e8..15 -> kb+16..23},
//             lanes 16-31 same M {K: kb+8..15, kb+24..31}.
__device__ __forceinline__ v16bf load_a_frag(const __bf16* base, int ld, int kb, int lane) {
  const int m = lane & 15, hf = lane >> 4;
  const __bf16* p = base + (size_t)m * ld + kb + 8 * hf;
  ABFrag u;
  u.h8[0] = *(const v8bf*)(p);
  u.h8[1] = *(const v8bf*)(p + 16);
  return u.v;
}
// B fragment (32x16 bf16). Weight stored row-major [N][K]; lane n = lane&15,
// lanes 0-15 hold K={kb+0..7, kb+16..23}, lanes 16-31 hold K={kb+8..15, kb+24..31}.
__device__ __forceinline__ v16bf load_b_frag(const __bf16* wbase, int ld, int kb, int lane) {
  const int n = lane & 15, hf = lane >> 4;
  const __bf16* p = wbase + (size_t)n * ld + kb + 8 * hf;
  ABFrag u;
  u.h8[0] = *(const v8bf*)(p);
  u.h8[1] = *(const v8bf*)(p + 16);
  return u.v;
}
__device__ __forceinline__ v8f wmma_bf16(v16bf a, v16bf b, v8f c) {
  return __builtin_amdgcn_wmma_f32_16x16x32_bf16(false, a, false, b, (short)0, c, false, false);
}

// --------------------------- K0: pack / convert ----------------------------

__global__ void init_state_kernel(__bf16* __restrict__ h, unsigned* __restrict__ bar) {
  const int i = blockIdx.x * blockDim.x + threadIdx.x;
  if (i == 0) *bar = 0u;
  if (i < 2 * B_SZ * H_SZ) h[i] = f2bf(0.f);
}

__global__ void pack_inp_kernel(const float* __restrict__ x, const float* __restrict__ feat,
                                const float* __restrict__ aux1, __bf16* __restrict__ inp) {
  const size_t total = (size_t)T_LEN * B_SZ * DPAD;
  for (size_t e = (size_t)blockIdx.x * blockDim.x + threadIdx.x; e < total;
       e += (size_t)gridDim.x * blockDim.x) {
    const size_t row = e / DPAD; const int c = (int)(e % DPAD);
    const int t = (int)(row >> 4), b = (int)(row & 15);
    float v = 0.f;
    if (c < 80)       v = feat[((size_t)b * T_LEN + t) * 80 + c];
    else if (c < 144) v = aux1[((size_t)b * T_LEN + t) * 64 + (c - 80)];
    else if (c < 147) v = x[((size_t)b * T_LEN + t) * 3 + (c - 144)];
    inp[e] = f2bf(v);
  }
}

__global__ void pack_wih_kernel(const float* __restrict__ w, __bf16* __restrict__ o) {
  const size_t total = (size_t)G3H * DPAD;
  for (size_t e = (size_t)blockIdx.x * blockDim.x + threadIdx.x; e < total;
       e += (size_t)gridDim.x * blockDim.x) {
    const int r = (int)(e / DPAD), c = (int)(e % DPAD);
    o[e] = f2bf(c < 147 ? w[(size_t)r * 147 + c] : 0.f);
  }
}

__global__ void cast_bf16_kernel(const float* __restrict__ s, __bf16* __restrict__ d, size_t n) {
  for (size_t e = (size_t)blockIdx.x * blockDim.x + threadIdx.x; e < n;
       e += (size_t)gridDim.x * blockDim.x)
    d[e] = f2bf(s[e]);
}

// ----------------------- K1: input projection GEMM -------------------------
// One block per timestep t; 8 waves x 24 n-tiles = 192 tiles = 3072 cols.

__global__ __launch_bounds__(256) void xp_gemm_kernel(
    const __bf16* __restrict__ inp, const __bf16* __restrict__ wih,
    const float* __restrict__ bih, float* __restrict__ xp) {
  const int t = blockIdx.x;
  const int wave = threadIdx.x >> 5, lane = threadIdx.x & 31;
  const int n = lane & 15, hf = lane >> 4;
  const __bf16* arow = inp + (size_t)t * B_SZ * DPAD;
  float* xpt = xp + (size_t)t * B_SZ * G3H;
  for (int j = 0; j < 24; ++j) {
    const int nt = wave * 24 + j;
    const __bf16* wrow = wih + (size_t)nt * 16 * DPAD;
    v8f acc = {};
#pragma unroll
    for (int k = 0; k < DPAD; k += 32) {
      v16bf a = load_a_frag(arow, DPAD, k, lane);
      v16bf b = load_b_frag(wrow, DPAD, k, lane);
      acc = wmma_bf16(a, b, acc);
    }
    const int col = nt * 16 + n;
    const float bias = bih[col];
#pragma unroll
    for (int i = 0; i < 8; ++i)
      xpt[(size_t)(i + 8 * hf) * G3H + col] = acc[i] + bias;
  }
}

// --------------------------- K2: persistent GRU ----------------------------
// 64 WGs x 96 threads. WG owns hidden units [16*wg, 16*wg+16); wave g in
// {0,1,2} computes gate-g rows as one 16x16 WMMA tile over K=1024, using two
// independent accumulator chains (k, k+512) to cut WMMA RAW latency.

__global__ __launch_bounds__(96) void gru_persistent_kernel(
    const __bf16* __restrict__ whh, const float* __restrict__ bhh,
    const float* __restrict__ xp, __bf16* __restrict__ hbuf,
    __bf16* __restrict__ hs, unsigned* __restrict__ bar) {
  __shared__ float sm_r[16][16];
  __shared__ float sm_z[16][16];
  const int u0   = blockIdx.x * 16;
  const int tid  = threadIdx.x;
  const int gate = tid >> 5;             // 0=r 1=z 2=n
  const int lane = tid & 31;
  const int n    = lane & 15;
  const int hf   = lane >> 4;
  const int grow = gate * H_SZ + u0 + n; // gate row = output column
  const float bias = bhh[grow];
  const __bf16* wrow = whh + (size_t)(gate * H_SZ + u0) * H_SZ;

  for (int t = 0; t < T_LEN; ++t) {
    const __bf16* h_in  = hbuf + (size_t)(t & 1) * (B_SZ * H_SZ);
    __bf16*       h_out = hbuf + (size_t)((t + 1) & 1) * (B_SZ * H_SZ);
    const float* xpt = xp + (size_t)t * (B_SZ * G3H);
    // hoist the x-projection gate values so they overlap the WMMA chain
    float xg[8];
#pragma unroll
    for (int i = 0; i < 8; ++i) xg[i] = xpt[(size_t)(i + 8 * hf) * G3H + grow];

    v8f acc0 = {}, acc1 = {};
#pragma unroll 2
    for (int k = 0; k < H_SZ / 2; k += 32) {
      __builtin_prefetch(wrow + (size_t)n * H_SZ + k + 32, 0, 0);
      v16bf a0 = load_a_frag(h_in, H_SZ, k, lane);
      v16bf b0 = load_b_frag(wrow, H_SZ, k, lane);
      acc0 = wmma_bf16(a0, b0, acc0);
      v16bf a1 = load_a_frag(h_in, H_SZ, k + H_SZ / 2, lane);
      v16bf b1 = load_b_frag(wrow, H_SZ, k + H_SZ / 2, lane);
      acc1 = wmma_bf16(a1, b1, acc1);
    }

    if (gate < 2) {
#pragma unroll
      for (int i = 0; i < 8; ++i) {
        const int m = i + 8 * hf;
        float g = xg[i] + acc0[i] + acc1[i] + bias;
        g = 1.f / (1.f + __expf(-g));
        if (gate == 0) sm_r[m][n] = g; else sm_z[m][n] = g;
      }
    }
    __syncthreads();
    if (gate == 2) {
#pragma unroll
      for (int i = 0; i < 8; ++i) {
        const int m = i + 8 * hf;
        const float r = sm_r[m][n], z = sm_z[m][n];
        const float ng = tanhf(xg[i] + r * (acc0[i] + acc1[i] + bias));
        const float hp = bf2f(h_in[m * H_SZ + u0 + n]);
        const __bf16 hb = f2bf((1.f - z) * ng + z * hp);
        h_out[m * H_SZ + u0 + n] = hb;
        hs[((size_t)t * B_SZ + m) * H_SZ + u0 + n] = hb;
      }
    }
    __threadfence();
    __syncthreads();
    if (tid == 0) {
      __hip_atomic_fetch_add(bar, 1u, __ATOMIC_RELEASE, __HIP_MEMORY_SCOPE_AGENT);
      const unsigned target = (unsigned)(t + 1) * GRU_WGS;
      while (__hip_atomic_load(bar, __ATOMIC_ACQUIRE, __HIP_MEMORY_SCOPE_AGENT) < target)
        __builtin_amdgcn_s_sleep(2);
    }
    __syncthreads();
  }
}

// ------------------------------ K3: fused head -----------------------------
// One block per t: A[16x576] staged in LDS (hs half via async-to-LDS if the
// toolchain exposes it, aux via VALU convert) -> relu GEMM -> o[16x1024] bf16
// in LDS -> GEMM -> logits[16x256] in LDS -> log_softmax -> out[b][t][256].

__global__ __launch_bounds__(256) void head_kernel(
    const __bf16* __restrict__ hs, const int hoff,
    const float* __restrict__ aux,
    const __bf16* __restrict__ wa, const float* __restrict__ ba,
    const __bf16* __restrict__ wb, const float* __restrict__ bb,
    float* __restrict__ out) {
  __shared__ __align__(64) char smraw[18432 + 32768];
  __bf16* a1   = (__bf16*)smraw;                    // [16][576]  (phase 1)
  __bf16* o1   = (__bf16*)(smraw + 18432);          // [16][1024]
  float*  lg   = (float*)smraw;                     // [16][256]  (reuses a1)
  float*  red  = (float*)(smraw + 16384);           // [16][16]
  float*  rowv = (float*)(smraw + 16384 + 1024);    // [16]

  const int t = blockIdx.x;
  const int tid = threadIdx.x;

#if defined(__HIP_DEVICE_COMPILE__) && \
    __has_builtin(__builtin_amdgcn_global_load_async_to_lds_b128)
  // Async stage of the hidden-state half: 16 rows x 512 bf16 = 1024 x 16B,
  // 4 chunks per thread, tracked on ASYNCcnt and overlapped with the aux fill.
  for (int it = 0; it < 4; ++it) {
    const int c2 = tid + 256 * it;
    const int row = c2 >> 6, chunk = c2 & 63;
    const __bf16* src = hs + ((size_t)t * B_SZ + row) * H_SZ + hoff + chunk * 8;
    __bf16* dst = a1 + row * KIN + chunk * 8;
    __builtin_amdgcn_global_load_async_to_lds_b128(
        (GLOBAL_AS v4i_t*)(unsigned long long)src, (LDS_AS v4i_t*)dst, 0, 0);
  }
  for (int e = tid; e < 16 * 64; e += 256) {        // aux columns 512..575
    const int m = e >> 6, c = e & 63;
    a1[m * KIN + 512 + c] = f2bf(aux[((size_t)m * T_LEN + t) * 64 + c]);
  }
#if __has_builtin(__builtin_amdgcn_s_wait_asynccnt)
  __builtin_amdgcn_s_wait_asynccnt(0);
#else
  asm volatile("s_wait_asynccnt 0x0" ::: "memory");
#endif
#else
  for (int e = tid; e < 16 * KIN; e += 256) {
    const int m = e / KIN, c = e % KIN;
    a1[e] = (c < 512) ? hs[((size_t)t * B_SZ + m) * H_SZ + hoff + c]
                      : f2bf(aux[((size_t)m * T_LEN + t) * 64 + (c - 512)]);
  }
#endif
  __syncthreads();

  const int wave = tid >> 5, lane = tid & 31;
  const int n = lane & 15, hf = lane >> 4;
  for (int j = 0; j < 8; ++j) {                       // GEMM1: 64 n-tiles
    const int nt = wave * 8 + j;
    const __bf16* wrow = wa + (size_t)nt * 16 * KIN;
    v8f acc = {};
#pragma unroll
    for (int k = 0; k < KIN; k += 32) {
      v16bf a = load_a_frag(a1, KIN, k, lane);
      v16bf b = load_b_frag(wrow, KIN, k, lane);
      acc = wmma_bf16(a, b, acc);
    }
    const int col = nt * 16 + n;
    const float bias = ba[col];
#pragma unroll
    for (int i = 0; i < 8; ++i)
      o1[(i + 8 * hf) * FC_SZ + col] = f2bf(fmaxf(acc[i] + bias, 0.f));
  }
  __syncthreads();
  for (int j = 0; j < 2; ++j) {                       // GEMM2: 16 n-tiles
    const int nt = wave * 2 + j;
    const __bf16* wrow = wb + (size_t)nt * 16 * FC_SZ;
    v8f acc = {};
#pragma unroll 4
    for (int k = 0; k < FC_SZ; k += 32) {
      v16bf a = load_a_frag(o1, FC_SZ, k, lane);
      v16bf b = load_b_frag(wrow, FC_SZ, k, lane);
      acc = wmma_bf16(a, b, acc);
    }
    const int col = nt * 16 + n;
    const float bias = bb[col];
#pragma unroll
    for (int i = 0; i < 8; ++i)
      lg[(i + 8 * hf) * NC_SZ + col] = acc[i] + bias;
  }
  __syncthreads();
  // log_softmax: 16 threads per row, 16 cols each
  const int r = tid >> 4, c0 = tid & 15;
  float mx = -3.4e38f;
  for (int j2 = 0; j2 < 16; ++j2) mx = fmaxf(mx, lg[r * NC_SZ + c0 + 16 * j2]);
  red[r * 16 + c0] = mx;
  __syncthreads();
  if (c0 == 0) {
    float m2 = red[r * 16];
    for (int i = 1; i < 16; ++i) m2 = fmaxf(m2, red[r * 16 + i]);
    rowv[r] = m2;
  }
  __syncthreads();
  const float rm = rowv[r];
  float s = 0.f;
  for (int j2 = 0; j2 < 16; ++j2) s += __expf(lg[r * NC_SZ + c0 + 16 * j2] - rm);
  red[r * 16 + c0] = s;
  __syncthreads();
  if (c0 == 0) {
    float s2 = 0.f;
    for (int i = 0; i < 16; ++i) s2 += red[r * 16 + i];
    rowv[r] = rm + __logf(s2);
  }
  __syncthreads();
  const float ls = rowv[r];
  for (int j2 = 0; j2 < 16; ++j2) {
    const int col = c0 + 16 * j2;
    out[((size_t)r * T_LEN + t) * NC_SZ + col] = lg[r * NC_SZ + col] - ls;
  }
}

// ------------------------------- launcher ----------------------------------

extern "C" void kernel_launch(void* const* d_in, const int* in_sizes, int n_in,
                              void* d_out, int out_size, void* d_ws, size_t ws_size,
                              hipStream_t stream) {
  const float* x    = (const float*)d_in[0];
  const float* feat = (const float*)d_in[1];
  const float* aux1 = (const float*)d_in[2];
  const float* aux2 = (const float*)d_in[3];
  const float* aux3 = (const float*)d_in[4];
  const float* w_ih = (const float*)d_in[5];
  const float* w_hh = (const float*)d_in[6];
  const float* b_ih = (const float*)d_in[7];
  const float* b_hh = (const float*)d_in[8];
  const float* w1 = (const float*)d_in[9];
  const float* b1 = (const float*)d_in[10];
  const float* w2 = (const float*)d_in[11];
  const float* b2 = (const float*)d_in[12];
  const float* w3 = (const float*)d_in[13];
  const float* b3 = (const float*)d_in[14];
  const float* w4 = (const float*)d_in[15];
  const float* b4 = (const float*)d_in[16];
  (void)in_sizes; (void)n_in; (void)out_size; (void)ws_size;

  char* ws = (char*)d_ws;
  size_t off = 0;
  auto take = [&](size_t bytes) { char* p = ws + off; off = (off + bytes + 255) & ~(size_t)255; return p; };
  unsigned* bar  = (unsigned*)take(4);
  __bf16* hbuf = (__bf16*)take((size_t)2 * B_SZ * H_SZ * 2);
  __bf16* inp  = (__bf16*)take((size_t)T_LEN * B_SZ * DPAD * 2);
  __bf16* wihB = (__bf16*)take((size_t)G3H * DPAD * 2);
  __bf16* whhB = (__bf16*)take((size_t)G3H * H_SZ * 2);
  __bf16* w1B  = (__bf16*)take((size_t)FC_SZ * KIN * 2);
  __bf16* w3B  = (__bf16*)take((size_t)FC_SZ * KIN * 2);
  __bf16* w2B  = (__bf16*)take((size_t)NC_SZ * FC_SZ * 2);
  __bf16* w4B  = (__bf16*)take((size_t)NC_SZ * FC_SZ * 2);
  float*  xp   = (float*)take((size_t)T_LEN * B_SZ * G3H * 4);
  __bf16* hs   = (__bf16*)take((size_t)T_LEN * B_SZ * H_SZ * 2);

  init_state_kernel<<<dim3(128), dim3(256), 0, stream>>>(hbuf, bar);
  pack_inp_kernel<<<dim3(4096), dim3(256), 0, stream>>>(x, feat, aux1, inp);
  pack_wih_kernel<<<dim3(512), dim3(256), 0, stream>>>(w_ih, wihB);
  cast_bf16_kernel<<<dim3(2048), dim3(256), 0, stream>>>(w_hh, whhB, (size_t)G3H * H_SZ);
  cast_bf16_kernel<<<dim3(512),  dim3(256), 0, stream>>>(w1, w1B, (size_t)FC_SZ * KIN);
  cast_bf16_kernel<<<dim3(512),  dim3(256), 0, stream>>>(w3, w3B, (size_t)FC_SZ * KIN);
  cast_bf16_kernel<<<dim3(256),  dim3(256), 0, stream>>>(w2, w2B, (size_t)NC_SZ * FC_SZ);
  cast_bf16_kernel<<<dim3(256),  dim3(256), 0, stream>>>(w4, w4B, (size_t)NC_SZ * FC_SZ);

  xp_gemm_kernel<<<dim3(T_LEN), dim3(256), 0, stream>>>(inp, wihB, b_ih, xp);
  gru_persistent_kernel<<<dim3(GRU_WGS), dim3(96), 0, stream>>>(whhB, b_hh, xp, hbuf, hs, bar);

  float* out = (float*)d_out;
  head_kernel<<<dim3(T_LEN), dim3(256), 0, stream>>>(hs, 0,   aux2, w1B, b1, w2B, b2, out);
  head_kernel<<<dim3(T_LEN), dim3(256), 0, stream>>>(hs, 512, aux3, w3B, b3, w4B, b4,
                                                     out + (size_t)B_SZ * T_LEN * NC_SZ);
}
